// MultiLatentAttention_89223650607640
// MI455X (gfx1250) — matrix operand order repaired and checked by reading, not compiled, not run on hardware
//
#include <hip/hip_runtime.h>
#include <hip/hip_bf16.h>

// ---------------------------------------------------------------------------
// MLA forward for MI455X (gfx1250), wave32, bf16 WMMA (v_wmma_f32_16x16x32_bf16)
// Flash-attention stages K/V tiles into LDS via GLOBAL_LOAD_ASYNC_TO_LDS_B128
// (ASYNCcnt) when the toolchain exposes the builtin; falls back to
// load+ds_store otherwise.
//
// B=2, S=2048, HID=128, NH=16, QK_D=V_D=128, Q_RANK=512, KV_RANK=256
// ---------------------------------------------------------------------------

#define BATCH   2
#define SEQ     2048
#define HID     128
#define NH      16
#define DHEAD   128
#define QRANK   512
#define KVRANK  256
#define ROWS    (BATCH * SEQ)          // 4096
#define QCOLS   (NH * DHEAD)           // 2048
#define KVCOLS  (NH * (DHEAD + DHEAD)) // 4096

typedef __attribute__((ext_vector_type(8)))  float  v8f;
typedef __bf16 bf16x8  __attribute__((ext_vector_type(8)));
typedef __bf16 bf16x16 __attribute__((ext_vector_type(16)));
typedef int vi4 __attribute__((vector_size(16)));   // matches async-LDS builtin param

#if defined(__has_builtin)
#if __has_builtin(__builtin_amdgcn_global_load_async_to_lds_b128)
#define HAS_ASYNC_LDS 1
#endif
#endif
#ifndef HAS_ASYNC_LDS
#define HAS_ASYNC_LDS 0
#endif

__device__ __forceinline__ unsigned short f32_to_bf16(float f) {
    unsigned int u = __builtin_bit_cast(unsigned int, f);
    unsigned int r = u + 0x7FFFu + ((u >> 16) & 1u);   // round-to-nearest-even
    return (unsigned short)(r >> 16);
}

__device__ __forceinline__ bf16x16 pack16(bf16x8 lo, bf16x8 hi) {
    bf16x16 r;
#pragma unroll
    for (int i = 0; i < 8; ++i) { r[i] = lo[i]; r[i + 8] = hi[i]; }
    return r;
}

// A fragment (16x32, MxK): lane holds row (lane&15), K (lane>>4)*16 .. +15
// contiguous.  Two 16-byte loads.
__device__ __forceinline__ bf16x16 load_a_frag(const unsigned short* __restrict__ p) {
    return pack16(*(const bf16x8*)p, *(const bf16x8*)(p + 8));
}

// B fragment: 16 contiguous bf16 per lane, as two 16-byte loads (works for
// 16B-aligned LDS rows where a single 32B vector deref would over-align).
__device__ __forceinline__ bf16x16 load_contig16(const unsigned short* __restrict__ p) {
    return pack16(*(const bf16x8*)p, *(const bf16x8*)(p + 8));
}

__device__ __forceinline__ v8f wmma_bf16(bf16x16 a, bf16x16 b, v8f c) {
    return __builtin_amdgcn_wmma_f32_16x16x32_bf16(false, a, false, b, (short)0, c,
                                                   false, false);
}

// one 16-byte global->LDS transfer per lane
__device__ __forceinline__ void stage16(const unsigned short* g, unsigned short* l) {
#if HAS_ASYNC_LDS
    __builtin_amdgcn_global_load_async_to_lds_b128(
        (__attribute__((address_space(1))) vi4*)g,
        (__attribute__((address_space(3))) vi4*)l, 0, 0);
#else
    *(bf16x8*)l = *(const bf16x8*)g;
#endif
}

__device__ __forceinline__ void stage_wait() {
#if HAS_ASYNC_LDS
#if defined(__has_builtin) && __has_builtin(__builtin_amdgcn_s_wait_asynccnt)
    __builtin_amdgcn_s_wait_asynccnt(0);
#else
    asm volatile("s_wait_asynccnt 0" ::: "memory");
#endif
#endif
}

// ---------------------------------------------------------------------------
// elementwise f32 -> bf16 copy
// ---------------------------------------------------------------------------
__global__ void k_f32_to_bf16(const float* __restrict__ in,
                              unsigned short* __restrict__ out, int n) {
    int t = blockIdx.x * blockDim.x + threadIdx.x;
    if (t < n) out[t] = f32_to_bf16(in[t]);
}

// in[K][N] (row-major)  ->  out[N][K] bf16 (transposed, row-major)
__global__ void k_convT(const float* __restrict__ in,
                        unsigned short* __restrict__ out, int Kd, int Nd) {
    int t = blockIdx.x * blockDim.x + threadIdx.x;
    if (t >= Kd * Nd) return;
    int k = t / Nd, n = t - k * Nd;
    out[(size_t)n * Kd + k] = f32_to_bf16(in[t]);
}

// kv buffer [ROWS][4096] -> vT[b][h][d][s]  (v part = cols 2048..4095)
__global__ void k_extract_vT(const unsigned short* __restrict__ kv,
                             unsigned short* __restrict__ vT) {
    int t = blockIdx.x * blockDim.x + threadIdx.x;   // < 2^23
    int s = t & (SEQ - 1);
    int d = (t >> 11) & (DHEAD - 1);
    int h = (t >> 18) & (NH - 1);
    int b = t >> 22;
    vT[t] = kv[((size_t)(b * SEQ + s)) * KVCOLS + QCOLS + h * DHEAD + d];
}

// ---------------------------------------------------------------------------
// Generic bf16 GEMM:  C[M x N] = A[M x K] * Bt[N x K]^T   (f32 accumulate)
// grid = (M/128, N/64), block = 256 (8 waves), wave tile = 32x32 (4 WMMA accs)
// ---------------------------------------------------------------------------
template <bool STORE_F32>
__global__ __launch_bounds__(256) void k_gemm_bf16(
    const unsigned short* __restrict__ A,
    const unsigned short* __restrict__ Bt,
    void* __restrict__ Cv, int N, int K) {
    const int lane = threadIdx.x & 31;
    const int wave = threadIdx.x >> 5;
    const int ln   = lane & 15;
    const int half = lane >> 4;
    const int r0 = blockIdx.x * 128 + (wave & 3) * 32;
    const int c0 = blockIdx.y * 64  + (wave >> 2) * 32;

    v8f acc[2][2] = {};
    for (int k0 = 0; k0 < K; k0 += 32) {
        bf16x16 af[2], bfr[2];
#pragma unroll
        for (int t = 0; t < 2; ++t)
            af[t] = load_a_frag(A + (size_t)(r0 + t * 16 + ln) * K + k0 + half * 16);
#pragma unroll
        for (int t = 0; t < 2; ++t)
            bfr[t] = *(const bf16x16*)(Bt + (size_t)(c0 + t * 16 + ln) * K +
                                       k0 + half * 16);
#pragma unroll
        for (int i = 0; i < 2; ++i)
#pragma unroll
            for (int j = 0; j < 2; ++j)
                acc[i][j] = wmma_bf16(af[i], bfr[j], acc[i][j]);
    }

#pragma unroll
    for (int i = 0; i < 2; ++i)
#pragma unroll
        for (int j = 0; j < 2; ++j) {
            const int col = c0 + j * 16 + ln;
#pragma unroll
            for (int e = 0; e < 8; ++e) {
                const int row = r0 + i * 16 + e + 8 * half;
                if (STORE_F32)
                    ((float*)Cv)[(size_t)row * N + col] = acc[i][j][e];
                else
                    ((unsigned short*)Cv)[(size_t)row * N + col] =
                        f32_to_bf16(acc[i][j][e]);
            }
        }
}

// ---------------------------------------------------------------------------
// Flash attention.  grid = (S/64, B*NH), block = 128 (4 waves).
// Wave w handles 16 query rows; a 64-key block is staged cooperatively
// into LDS (async-to-LDS on gfx1250), then all waves consume it.
//   q  : [ROWS][2048] bf16   (row = b*S+s, col = h*128+d)
//   kv : [ROWS][4096] bf16   (k part = cols 0..2047)
//   vT : [B*NH*128][S] bf16
//   ao : [ROWS][2048] bf16
// ---------------------------------------------------------------------------
#define PLD  72   // padded P row stride (bf16 elems)
#define KSTR 136  // padded K-tile row stride: 272B = 17*16B -> bank-staggered
#define VSTR 72   // padded V-tile row stride: 144B =  9*16B

__global__ __launch_bounds__(128) void k_mla_flash(
    const unsigned short* __restrict__ q,
    const unsigned short* __restrict__ kv,
    const unsigned short* __restrict__ vT,
    unsigned short* __restrict__ ao) {
    __shared__ __align__(16) unsigned short kls[64 * KSTR];       // 17408 B
    __shared__ __align__(16) unsigned short vls[128 * VSTR];      // 18432 B
    __shared__ __align__(16) unsigned short plds[4 * 16 * PLD];   //  9216 B

    const int tid  = threadIdx.x;
    const int lane = tid & 31;
    const int wave = tid >> 5;
    const int ln   = lane & 15;
    const int half = lane >> 4;
    const int b = (int)blockIdx.y >> 4;
    const int h = (int)blockIdx.y & 15;
    const int qm0 = (int)blockIdx.x * 64 + wave * 16;
    const float scale = 0.088388347648318447f;   // 1/sqrt(128)

    // -------- load Q fragments (D=128 -> 4 fragments of 16x32) --------
    bf16x16 qf[4];
    {
        const size_t qbase =
            (size_t)(b * SEQ + qm0 + ln) * QCOLS + h * DHEAD + half * 16;
#pragma unroll
        for (int kf = 0; kf < 4; ++kf)
            qf[kf] = load_a_frag(q + qbase + kf * 32);
    }

    v8f   acc[8] = {};                 // 16 rows x 128 cols (f32)
    float mrow[8], lrow[8];
#pragma unroll
    for (int i = 0; i < 8; ++i) { mrow[i] = -1e30f; lrow[i] = 0.0f; }

    unsigned short* pw = plds + wave * 16 * PLD;
    const size_t kgbase = (size_t)(b * SEQ) * KVCOLS + h * DHEAD;      // + (kb+r)*KVCOLS
    const size_t vgbase = (size_t)((b * NH + h) * DHEAD) * SEQ;        // + d*SEQ + kb

    for (int kb = 0; kb < SEQ; kb += 64) {
        // ---- cooperative staging: K tile 64x128, V tile 128x64 (bf16) ----
#pragma unroll
        for (int i = 0; i < 8; ++i) {               // K: 1024 x 16B chunks
            const int c  = tid + i * 128;
            const int r  = c >> 4, cc = c & 15;
            stage16(kv + kgbase + (size_t)(kb + r) * KVCOLS + cc * 8,
                    kls + r * KSTR + cc * 8);
        }
#pragma unroll
        for (int i = 0; i < 8; ++i) {               // V: 1024 x 16B chunks
            const int c  = tid + i * 128;
            const int d  = c >> 3, cc = c & 7;
            stage16(vT + vgbase + (size_t)d * SEQ + kb + cc * 8,
                    vls + d * VSTR + cc * 8);
        }
        stage_wait();
        __syncthreads();

        // ---- scores S = scale * Q K^T   (16 x 64 tile, 16 WMMAs) ----
        v8f sacc[4] = {};
#pragma unroll
        for (int nt = 0; nt < 4; ++nt) {
            const unsigned short* kp = kls + (nt * 16 + ln) * KSTR + half * 16;
#pragma unroll
            for (int kf = 0; kf < 4; ++kf) {
                bf16x16 kfr = load_contig16(kp + kf * 32);
                sacc[nt] = wmma_bf16(qf[kf], kfr, sacc[nt]);
            }
        }

        // ---- online softmax (rows live across 16-lane halves) ----
        float bmax[8];
#pragma unroll
        for (int i = 0; i < 8; ++i) bmax[i] = -1e30f;
#pragma unroll
        for (int nt = 0; nt < 4; ++nt)
#pragma unroll
            for (int i = 0; i < 8; ++i) {
                sacc[nt][i] *= scale;
                bmax[i] = fmaxf(bmax[i], sacc[nt][i]);
            }
#pragma unroll
        for (int m = 1; m < 16; m <<= 1)
#pragma unroll
            for (int i = 0; i < 8; ++i)
                bmax[i] = fmaxf(bmax[i], __shfl_xor(bmax[i], m, 32));

        float corr[8], bsum[8];
#pragma unroll
        for (int i = 0; i < 8; ++i) {
            float nm = fmaxf(mrow[i], bmax[i]);
            corr[i] = __expf(mrow[i] - nm);
            mrow[i] = nm;
            bsum[i] = 0.0f;
        }
#pragma unroll
        for (int nt = 0; nt < 4; ++nt)
#pragma unroll
            for (int i = 0; i < 8; ++i) {
                float p = __expf(sacc[nt][i] - mrow[i]);
                sacc[nt][i] = p;
                bsum[i] += p;
            }
#pragma unroll
        for (int m = 1; m < 16; m <<= 1)
#pragma unroll
            for (int i = 0; i < 8; ++i)
                bsum[i] += __shfl_xor(bsum[i], m, 32);
#pragma unroll
        for (int i = 0; i < 8; ++i) lrow[i] = lrow[i] * corr[i] + bsum[i];
#pragma unroll
        for (int ot = 0; ot < 8; ++ot)
#pragma unroll
            for (int i = 0; i < 8; ++i) acc[ot][i] *= corr[i];

        // ---- P (C-layout) -> per-wave LDS -> A-layout bf16 fragments ----
#pragma unroll
        for (int nt = 0; nt < 4; ++nt)
#pragma unroll
            for (int i = 0; i < 8; ++i)
                pw[(i + 8 * half) * PLD + nt * 16 + ln] =
                    f32_to_bf16(sacc[nt][i]);
        // per-wave region; same-wave LDS ops are processed in order.

        bf16x16 pf[2];
#pragma unroll
        for (int kf = 0; kf < 2; ++kf)
            pf[kf] = load_a_frag(pw + ln * PLD + kf * 32 + half * 16);

        // ---- O += P V   (16 x 128, 16 WMMAs) ----
#pragma unroll
        for (int ot = 0; ot < 8; ++ot) {
            const unsigned short* vp = vls + (ot * 16 + ln) * VSTR + half * 16;
#pragma unroll
            for (int kf = 0; kf < 2; ++kf) {
                bf16x16 vfr = load_contig16(vp + kf * 32);
                acc[ot] = wmma_bf16(pf[kf], vfr, acc[ot]);
            }
        }
        __syncthreads();   // K/V tiles consumed; safe to restage next block
    }

    // ---- normalize and store ----
#pragma unroll
    for (int ot = 0; ot < 8; ++ot) {
        const int col = h * DHEAD + ot * 16 + ln;
#pragma unroll
        for (int i = 0; i < 8; ++i) {
            const int row = qm0 + i + 8 * half;
            float o = acc[ot][i] / lrow[i];
            ao[(size_t)(b * SEQ + row) * QCOLS + col] = f32_to_bf16(o);
        }
    }
}

// ---------------------------------------------------------------------------
// host-side launch
// ---------------------------------------------------------------------------
static inline size_t align256(size_t x) { return (x + 255) & ~(size_t)255; }

extern "C" void kernel_launch(void* const* d_in, const int* in_sizes, int n_in,
                              void* d_out, int out_size, void* d_ws, size_t ws_size,
                              hipStream_t stream) {
    const float* x    = (const float*)d_in[0];
    const float* Wqd  = (const float*)d_in[1];   // [128][512]
    const float* Wqu  = (const float*)d_in[2];   // [512][2048]
    const float* Wkvd = (const float*)d_in[3];   // [128][256]
    const float* Wkvu = (const float*)d_in[4];   // [256][4096]
    const float* Wo   = (const float*)d_in[5];   // [2048][128]

    char* ws = (char*)d_ws;
    size_t off = 0;
    auto alloc = [&](size_t bytes) { void* p = ws + off; off += align256(bytes); return p; };

    unsigned short* xb     = (unsigned short*)alloc((size_t)ROWS * HID * 2);
    unsigned short* WqdT   = (unsigned short*)alloc((size_t)QRANK * HID * 2);
    unsigned short* WquT   = (unsigned short*)alloc((size_t)QCOLS * QRANK * 2);
    unsigned short* WkvdT  = (unsigned short*)alloc((size_t)KVRANK * HID * 2);
    unsigned short* WkvuT  = (unsigned short*)alloc((size_t)KVCOLS * KVRANK * 2);
    unsigned short* WoT    = (unsigned short*)alloc((size_t)HID * QCOLS * 2);
    unsigned short* qlat   = (unsigned short*)alloc((size_t)ROWS * QRANK * 2);
    unsigned short* qbuf   = (unsigned short*)alloc((size_t)ROWS * QCOLS * 2);
    unsigned short* kvlat  = (unsigned short*)alloc((size_t)ROWS * KVRANK * 2);
    unsigned short* kvbuf  = (unsigned short*)alloc((size_t)ROWS * KVCOLS * 2);
    unsigned short* vT     = (unsigned short*)alloc((size_t)BATCH * NH * DHEAD * SEQ * 2);
    unsigned short* ao     = (unsigned short*)alloc((size_t)ROWS * QCOLS * 2);
    (void)ws_size; (void)in_sizes; (void)n_in; (void)out_size;

    const int T = 256;
    // conversions
    k_f32_to_bf16<<<(ROWS * HID + T - 1) / T, T, 0, stream>>>(x, xb, ROWS * HID);
    k_convT<<<(HID * QRANK + T - 1) / T, T, 0, stream>>>(Wqd, WqdT, HID, QRANK);
    k_convT<<<(QRANK * QCOLS + T - 1) / T, T, 0, stream>>>(Wqu, WquT, QRANK, QCOLS);
    k_convT<<<(HID * KVRANK + T - 1) / T, T, 0, stream>>>(Wkvd, WkvdT, HID, KVRANK);
    k_convT<<<(KVRANK * KVCOLS + T - 1) / T, T, 0, stream>>>(Wkvu, WkvuT, KVRANK, KVCOLS);
    k_convT<<<(QCOLS * HID + T - 1) / T, T, 0, stream>>>(Wo, WoT, QCOLS, HID);

    // projection GEMM chain (grid = M/128 x N/64)
    k_gemm_bf16<false><<<dim3(ROWS / 128, QRANK / 64), T, 0, stream>>>(xb, WqdT, qlat, QRANK, HID);
    k_gemm_bf16<false><<<dim3(ROWS / 128, QCOLS / 64), T, 0, stream>>>(qlat, WquT, qbuf, QCOLS, QRANK);
    k_gemm_bf16<false><<<dim3(ROWS / 128, KVRANK / 64), T, 0, stream>>>(xb, WkvdT, kvlat, KVRANK, HID);
    k_gemm_bf16<false><<<dim3(ROWS / 128, KVCOLS / 64), T, 0, stream>>>(kvlat, WkvuT, kvbuf, KVCOLS, KVRANK);

    // v transpose for the PV matmul
    k_extract_vT<<<(BATCH * NH * DHEAD * SEQ) / T, T, 0, stream>>>(kvbuf, vT);

    // flash attention
    k_mla_flash<<<dim3(SEQ / 64, BATCH * NH), 128, 0, stream>>>(qbuf, kvbuf, vT, ao);

    // output projection -> f32 result
    k_gemm_bf16<true><<<dim3(ROWS / 128, HID / 64), T, 0, stream>>>(ao, WoT, d_out, HID, QCOLS);
}
